// TransBlockV2_64372969832682
// MI455X (gfx1250) — matrix operand order repaired and checked by reading, not compiled, not run on hardware
//
#include <hip/hip_runtime.h>
#include <hip/hip_bf16.h>
#include <stdint.h>

typedef __bf16 bf16;
typedef __attribute__((ext_vector_type(16))) __bf16 v16bf;
typedef __attribute__((ext_vector_type(8)))  __bf16 v8bf;
typedef __attribute__((ext_vector_type(8)))  float  v8f;

#define NN 10000
#define EE 50000

__device__ __constant__ int c_EXPAND[25] =
    {0,1,1,1,2,2,2,2,2,3,3,3,3,3,3,3,4,4,4,4,4,4,4,4,4};

__device__ inline float siluf(float x) { return x / (1.f + __expf(-x)); }

// ordered-uint encoding for float atomic max
__device__ inline unsigned fenc(float f) {
  unsigned u = __float_as_uint(f);
  return (u & 0x80000000u) ? ~u : (u | 0x80000000u);
}
__device__ inline float fdec(unsigned k) {
  unsigned u = (k & 0x80000000u) ? (k & 0x7fffffffu) : ~k;
  return __uint_as_float(u);
}

// ---------------- WMMA fragment helpers (wave32, ISA 7.12.2) ----------------
__device__ inline v16bf cat8(v8bf lo, v8bf hi) {
  return __builtin_shufflevector(lo, hi, 0, 1, 2, 3, 4, 5, 6, 7, 8, 9, 10, 11,
                                 12, 13, 14, 15);
}
// A 16x32 tile from row-major storage (ld bf16, 16B-aligned rows)
__device__ inline v16bf load_a_rm(const bf16* A, int ld, int lane) {
  const int row  = lane & 15;
  const int koff = (lane & 16) ? 16 : 0;
  const bf16* p = A + row * ld + koff;
  v8bf lo = *(const v8bf*)p;
  v8bf hi = *(const v8bf*)(p + 8);
  return cat8(lo, hi);
}
// B matrices stored swizzled in 16col x 16k blocks, k contiguous per column:
// element (k,c) at ((c>>4)*KB + (k>>4))*256 + (c&15)*16 + (k&15), KB = K/16.
__device__ inline v16bf load_b_swz(const bf16* M, int KB, int ct, int kc,
                                   int lane) {
  const int cc = lane & 15;
  const int kh = kc * 2 + ((lane >> 4) & 1);
  const bf16* p = M + (ct * KB + kh) * 256 + cc * 16;
  v8bf lo = *(const v8bf*)p;
  v8bf hi = *(const v8bf*)(p + 8);
  return cat8(lo, hi);
}
// D tile (16x16 f32) -> swizzled bf16 B-matrix: one b128 store per lane.
__device__ inline void store_d_swz(bf16* M, int KB, int ct, int mt, int lane,
                                   v8f vals) {
  const int cc   = lane & 15;
  const int drow = (lane & 16) ? 8 : 0;
  v8bf pk;
#pragma unroll
  for (int r = 0; r < 8; ++r) pk[r] = (bf16)vals[r];
  *(v8bf*)(M + (ct * KB + mt) * 256 + cc * 16 + drow) = pk;
}
__device__ inline v8f wmma_bf(v16bf a, v16bf b, v8f c) {
  return __builtin_amdgcn_wmma_f32_16x16x32_bf16(false, a, false, b, (short)0, c,
                                                 false, false);
}

// ---------------- async global->LDS (CDNA5 ASYNCcnt path) ----------------
__device__ inline void async_b128(bf16* lds_dst, const bf16* gsrc) {
  unsigned ldsoff = (unsigned)(uintptr_t)lds_dst;
  unsigned long long ga = (unsigned long long)(uintptr_t)gsrc;
  asm volatile("global_load_async_to_lds_b128 %0, %1, off" ::"v"(ldsoff), "v"(ga)
               : "memory");
}
__device__ inline void wait_async0() {
  asm volatile("s_wait_asynccnt 0x0" ::: "memory");
}

// ---------------- K0: weight prep ----------------
__global__ void k_prep(const float* W1, const float* tg, const float* fg,
                       const float* proj, const float* f1, const float* f2,
                       const float* rw1, const float* rw2, const float* rw3,
                       bf16* W1b, bf16* tgb, bf16* fgb, bf16* projb, bf16* f1b,
                       bf16* f2b, bf16* rw1b, bf16* rw2b, bf16* rw3b) {
  int t0 = blockIdx.x * blockDim.x + threadIdx.x;
  int st = gridDim.x * blockDim.x;
  // B-operand weights -> swizzled
  for (int i = t0; i < 128 * 64; i += st) {
    int k = i >> 6, c = i & 63;
    W1b[((c >> 4) * 8 + (k >> 4)) * 256 + (c & 15) * 16 + (k & 15)] = (bf16)W1[i];
  }
  for (int i = t0; i < 64 * 64; i += st) {
    int k = i >> 6, c = i & 63;
    projb[((c >> 4) * 4 + (k >> 4)) * 256 + (c & 15) * 16 + (k & 15)] = (bf16)proj[i];
  }
  for (int i = t0; i < 64 * 128; i += st) {
    int k = i >> 7, c = i & 127;
    f1b[((c >> 4) * 4 + (k >> 4)) * 256 + (c & 15) * 16 + (k & 15)] = (bf16)f1[i];
  }
  for (int i = t0; i < 128 * 64; i += st) {
    int k = i >> 6, c = i & 63;
    f2b[((c >> 4) * 8 + (k >> 4)) * 256 + (c & 15) * 16 + (k & 15)] = (bf16)f2[i];
  }
  for (int i = t0; i < 96 * 64; i += st) {
    int k = i >> 6, c = i & 63;
    rw1b[((c >> 4) * 6 + (k >> 4)) * 256 + (c & 15) * 16 + (k & 15)] = (bf16)rw1[i];
  }
  for (int i = t0; i < 64 * 64; i += st) {
    int k = i >> 6, c = i & 63;
    rw2b[((c >> 4) * 4 + (k >> 4)) * 256 + (c & 15) * 16 + (k & 15)] = (bf16)rw2[i];
  }
  for (int i = t0; i < 64 * 640; i += st) {
    int k = i / 640, c = i % 640;
    rw3b[((c >> 4) * 4 + (k >> 4)) * 256 + (c & 15) * 16 + (k & 15)] = (bf16)rw3[i];
  }
  // A-operand weights -> row-major zero-padded
  for (int i = t0; i < 48 * 32; i += st) {
    int g = i >> 5, j = i & 31;
    tgb[i] = (g < 36 && j < 25) ? (bf16)tg[g * 25 + j] : (bf16)0.f;
  }
  for (int i = t0; i < 32 * 64; i += st) {
    int r = i >> 6, g = i & 63;
    fgb[i] = (r < 25 && g < 36) ? (bf16)fg[r * 36 + g] : (bf16)0.f;
  }
}

// ---------------- K1: rmsnorm -> transposed padded hbT[n][c][32] ----------------
__global__ void k_rms1(const float* x, const float* gamma, bf16* hbT) {
  const int n = blockIdx.x, c = threadIdx.x;  // blockDim = 64
  float s = 0.f;
  for (int i = 0; i < 25; ++i) {
    float v = x[(size_t)n * 1600 + i * 64 + c];
    s += v * v;
  }
  float inv = rsqrtf(s / 25.f + 1e-6f) * gamma[c];
  v8bf buf[4];
  bf16* bb = (bf16*)buf;
  for (int i = 0; i < 25; ++i)
    bb[i] = (bf16)(x[(size_t)n * 1600 + i * 64 + c] * inv);
  for (int i = 25; i < 32; ++i) bb[i] = (bf16)0.f;
  v8bf* dst = (v8bf*)(hbT + (size_t)n * 2048 + c * 32);
#pragma unroll
  for (int j = 0; j < 4; ++j) dst[j] = buf[j];
}

// ---------------- K2: batched radial MLP, 16 edges per wave (WMMA) ----------------
__global__ void __launch_bounds__(32)
k_rad(const int* ei, const int* anum, const float* edist, const float* stab,
      const float* ttab, const bf16* rw1b, const float* rb1, const bf16* rw2b,
      const float* rb2, const bf16* rw3b, const float* rb3, bf16* radb) {
  const int e0 = blockIdx.x * 16, lane = threadIdx.x;
  __shared__ __align__(16) bf16 sEf[16 * 96];
  __shared__ __align__(16) bf16 sL1[16 * 64];
  __shared__ __align__(16) bf16 sL2[16 * 64];
  for (int q = lane; q < 16 * 96; q += 32) {
    int i = q / 96, j = q % 96;
    int e = e0 + i;
    float v;
    if (j < 32)      v = edist[(size_t)e * 32 + j];
    else if (j < 64) v = stab[anum[ei[e]] * 32 + (j - 32)];
    else             v = ttab[anum[ei[EE + e]] * 32 + (j - 64)];
    sEf[q] = (bf16)v;
  }
  __syncthreads();
  const int drow = (lane & 16) ? 8 : 0, dcol = lane & 15;
  // layer1: (16x96)@(96x64), +b1, silu
  for (int nt = 0; nt < 4; ++nt) {
    v8f acc = {};
    for (int kt = 0; kt < 3; ++kt) {
      v16bf A = load_a_rm(sEf + kt * 32, 96, lane);
      v16bf B = load_b_swz(rw1b, 6, nt, kt, lane);
      acc = wmma_bf(A, B, acc);
    }
#pragma unroll
    for (int r = 0; r < 8; ++r) {
      int row = drow + r, gcol = nt * 16 + dcol;
      sL1[row * 64 + gcol] = (bf16)siluf(acc[r] + rb1[gcol]);
    }
  }
  __syncthreads();
  // layer2: (16x64)@(64x64), +b2, silu
  for (int nt = 0; nt < 4; ++nt) {
    v8f acc = {};
    for (int kt = 0; kt < 2; ++kt) {
      v16bf A = load_a_rm(sL1 + kt * 32, 64, lane);
      v16bf B = load_b_swz(rw2b, 4, nt, kt, lane);
      acc = wmma_bf(A, B, acc);
    }
#pragma unroll
    for (int r = 0; r < 8; ++r) {
      int row = drow + r, gcol = nt * 16 + dcol;
      sL2[row * 64 + gcol] = (bf16)siluf(acc[r] + rb2[gcol]);
    }
  }
  __syncthreads();
  // layer3: (16x64)@(64x640) + b3 -> radb
  for (int nt = 0; nt < 40; ++nt) {
    v8f acc = {};
    for (int kt = 0; kt < 2; ++kt) {
      v16bf A = load_a_rm(sL2 + kt * 32, 64, lane);
      v16bf B = load_b_swz(rw3b, 4, nt, kt, lane);
      acc = wmma_bf(A, B, acc);
    }
#pragma unroll
    for (int r = 0; r < 8; ++r) {
      int row = drow + r, gcol = nt * 16 + dcol;
      radb[(size_t)(e0 + row) * 640 + gcol] = (bf16)(acc[r] + rb3[gcol]);
    }
  }
}

// ---------------- K3: pass A — alpha + gate, segment max ----------------
__global__ void __launch_bounds__(32)
k_alpha(const int* ei, const float* wig, const float* W0e, const float* adot,
        const bf16* hbT, const bf16* radb, float* gate, float* alpha,
        unsigned* amaxk) {
  const int e = blockIdx.x, lane = threadIdx.x;
  __shared__ float xr[128], af[128];
  const int src = ei[e], dst = ei[EE + e];
  const float* w0 = wig + (size_t)e * 625;
  for (int cc = 0; cc < 4; ++cc) {
    int c = lane + cc * 32;
    float r = (float)radb[(size_t)e * 640 + c];  // l=0 slice
    const bf16* hrow = (c < 64) ? hbT + (size_t)src * 2048 + c * 32
                                : hbT + (size_t)dst * 2048 + (c - 64) * 32;
    float s = 0.f;
    for (int i = 0; i < 25; ++i) s += w0[i] * (float)hrow[i];
    xr[c] = s * r;
  }
  __syncthreads();
  for (int oo = 0; oo < 6; ++oo) {
    int o = lane + oo * 32;
    float s = 0.f;
    for (int c = 0; c < 128; ++c) s += xr[c] * W0e[c * 192 + o];
    if (o < 128) af[o] = s;
    else         gate[(size_t)e * 64 + (o - 128)] = s;
  }
  __syncthreads();
  if (lane < 8) {
    float s = 0.f;
    for (int a = 0; a < 16; ++a) s += siluf(af[lane * 16 + a]) * adot[lane * 16 + a];
    alpha[(size_t)e * 8 + lane] = s;
    atomicMax(&amaxk[(size_t)dst * 8 + lane], fenc(s));
  }
}

// ---------------- K4: pass B — segment sum of exp ----------------
__global__ void k_softsum(const int* ei, const float* alpha, const unsigned* amaxk,
                          float* asum) {
  int idx = blockIdx.x * blockDim.x + threadIdx.x;
  if (idx >= EE * 8) return;
  int e = idx >> 3, hd = idx & 7;
  int dst = ei[EE + e];
  float a = __expf(alpha[idx] - fdec(amaxk[dst * 8 + hd]));
  atomicAdd(&asum[dst * 8 + hd], a);
}

// ---------------- K5: pass C — full edge pipeline (WMMA) ----------------
__global__ void __launch_bounds__(32)
k_edge(const int* ei, const float* wig, const bf16* hbT, const float* gate,
       const float* alpha, const unsigned* amaxk, const float* asum,
       const bf16* radb, const bf16* W1b, const bf16* tgb, const bf16* fgb,
       float* agg) {
  const int e = blockIdx.x, lane = threadIdx.x;
  __shared__ __align__(16) bf16 sWig[32 * 32];
  __shared__ __align__(16) bf16 sWigT[32 * 32];
  __shared__ __align__(16) bf16 sXe[32 * 128];    // swz, KB=2
  __shared__ __align__(16) bf16 sXwr[32 * 128];   // row-major (A operand)
  __shared__ __align__(16) bf16 sMsg[32 * 64];    // swz, KB=2
  __shared__ __align__(16) bf16 sGrid[64 * 64];   // swz, KB=4
  __shared__ __align__(16) bf16 sMsg2[32 * 64];   // swz, KB=2
  __shared__ float sRad[5 * 128];
  __shared__ float sA[8];
  const int src = ei[e], dst = ei[EE + e];

  // async fill of swizzled xe from transposed h
  for (int q = lane; q < 512; q += 32) {
    int c = q >> 2, kh = (q >> 1) & 1, part = q & 1;
    const bf16* g = (c < 64)
                        ? hbT + (size_t)src * 2048 + c * 32 + kh * 16 + part * 8
                        : hbT + (size_t)dst * 2048 + (c - 64) * 32 + kh * 16 + part * 8;
    bf16* d = sXe + ((c >> 4) * 2 + kh) * 256 + (c & 15) * 16 + part * 8;
    async_b128(d, g);
  }

  // wigner + transpose into LDS (A operands, row-major)
  const float* we = wig + (size_t)e * 625;
  for (int t = lane; t < 32 * 32; t += 32) {
    int i = t >> 5, j = t & 31;
    bf16 v = (i < 25 && j < 25) ? (bf16)we[i * 25 + j] : (bf16)0.f;
    sWig[i * 32 + j]  = v;
    sWigT[j * 32 + i] = v;
  }
  // radial (precomputed)
  for (int o = lane; o < 640; o += 32)
    sRad[o] = (float)radb[(size_t)e * 640 + o];
  if (lane < 8) {
    float a = __expf(alpha[(size_t)e * 8 + lane] - fdec(amaxk[(size_t)dst * 8 + lane]));
    sA[lane] = a / (asum[(size_t)dst * 8 + lane] + 1e-8f);
  }
  // zero kh=3 blocks of sGrid (K padded 36->64)
  for (int q = lane; q < 128; q += 32) {
    int ct = q >> 5, w = q & 31;
    *(v8bf*)(sGrid + (ct * 4 + 3) * 256 + w * 8) = (v8bf)(bf16)0.f;
  }
  wait_async0();
  __syncthreads();

  const int drow = (lane & 16) ? 8 : 0;
  const int dcol = lane & 15;
  v8f zero = {};

  // xw = wigner @ xe ; * radial -> sXwr (row-major bf16, A operand)
  for (int mt = 0; mt < 2; ++mt) {
    v16bf A = load_a_rm(sWig + mt * 16 * 32, 32, lane);
    for (int nt = 0; nt < 8; ++nt) {
      v16bf B = load_b_swz(sXe, 2, nt, 0, lane);
      v8f D = wmma_bf(A, B, zero);
#pragma unroll
      for (int r = 0; r < 8; ++r) {
        int grow = mt * 16 + drow + r, gcol = nt * 16 + dcol;
        float val = (grow < 25) ? D[r] * sRad[c_EXPAND[grow] * 128 + gcol] : 0.f;
        sXwr[grow * 128 + gcol] = (bf16)val;
      }
    }
  }
  __syncthreads();
  // msg = xwr @ W1 -> sMsg (swz)
  for (int mt = 0; mt < 2; ++mt)
    for (int nt = 0; nt < 4; ++nt) {
      v8f acc = {};
      for (int kt = 0; kt < 4; ++kt) {
        v16bf A = load_a_rm(sXwr + mt * 16 * 128 + kt * 32, 128, lane);
        v16bf B = load_b_swz(W1b, 8, nt, kt, lane);
        acc = wmma_bf(A, B, acc);
      }
      store_d_swz(sMsg, 2, nt, mt, lane, acc);
    }
  __syncthreads();
  // grid = to_grid @ msg ; silu -> sGrid (swz)
  for (int mt = 0; mt < 3; ++mt) {
    v16bf A = load_a_rm(tgb + mt * 16 * 32, 32, lane);
    for (int nt = 0; nt < 4; ++nt) {
      v16bf B = load_b_swz(sMsg, 2, nt, 0, lane);
      v8f D = wmma_bf(A, B, zero);
      v8f t;
#pragma unroll
      for (int r = 0; r < 8; ++r) t[r] = siluf(D[r]);
      store_d_swz(sGrid, 4, nt, mt, lane, t);
    }
  }
  __syncthreads();
  // xg = from_grid @ grid ; row0 = silu(gate) ; * attention -> sMsg2 (swz)
  for (int mt = 0; mt < 2; ++mt)
    for (int nt = 0; nt < 4; ++nt) {
      v8f acc = {};
      for (int kt = 0; kt < 2; ++kt) {
        v16bf A = load_a_rm(fgb + mt * 16 * 64 + kt * 32, 64, lane);
        v16bf B = load_b_swz(sGrid, 4, nt, kt, lane);
        acc = wmma_bf(A, B, acc);
      }
      v8f t;
#pragma unroll
      for (int r = 0; r < 8; ++r) {
        int grow = mt * 16 + drow + r, gcol = nt * 16 + dcol;
        float v;
        if (grow == 0)      v = siluf(gate[(size_t)e * 64 + gcol]);
        else if (grow < 25) v = acc[r];
        else                v = 0.f;
        t[r] = v * sA[gcol >> 3];
      }
      store_d_swz(sMsg2, 2, nt, mt, lane, t);
    }
  __syncthreads();
  // out = wigner^T @ msg2 -> atomic agg
  for (int mt = 0; mt < 2; ++mt) {
    v16bf A = load_a_rm(sWigT + mt * 16 * 32, 32, lane);
    for (int nt = 0; nt < 4; ++nt) {
      v16bf B = load_b_swz(sMsg2, 2, nt, 0, lane);
      v8f D = wmma_bf(A, B, zero);
#pragma unroll
      for (int r = 0; r < 8; ++r) {
        int grow = mt * 16 + drow + r, gcol = nt * 16 + dcol;
        if (grow < 25)
          atomicAdd(&agg[(size_t)dst * 1600 + grow * 64 + gcol], D[r]);
      }
    }
  }
}

// ---------------- K6: xnew = x + agg@proj ; rmsnorm2 ; gate ----------------
__global__ void __launch_bounds__(32)
k_node_up(const float* x, const float* agg, const bf16* projb, const float* gamma2,
          const float* glw, const float* glb, float* xnew, bf16* h2b,
          float* gaten) {
  const int n = blockIdx.x, lane = threadIdx.x;
  __shared__ __align__(16) bf16 sAg[32 * 64];  // row-major A
  __shared__ __align__(16) float sX[25 * 64];
  __shared__ float sInv[64];
  for (int q = lane; q < 25 * 16; q += 32) {  // 4-float chunks
    int row = q >> 4, part = q & 15;
    const float* g = agg + (size_t)n * 1600 + row * 64 + part * 4;
    bf16* d = sAg + row * 64 + part * 4;
    d[0] = (bf16)g[0]; d[1] = (bf16)g[1]; d[2] = (bf16)g[2]; d[3] = (bf16)g[3];
  }
  for (int q = lane; q < 7 * 8; q += 32)
    *(v8bf*)(sAg + 25 * 64 + q * 8) = (v8bf)(bf16)0.f;
  __syncthreads();
  const int drow = (lane & 16) ? 8 : 0, dcol = lane & 15;
  for (int mt = 0; mt < 2; ++mt)
    for (int nt = 0; nt < 4; ++nt) {
      v8f acc = {};
      for (int kt = 0; kt < 2; ++kt) {
        v16bf A = load_a_rm(sAg + mt * 16 * 64 + kt * 32, 64, lane);
        v16bf B = load_b_swz(projb, 4, nt, kt, lane);
        acc = wmma_bf(A, B, acc);
      }
#pragma unroll
      for (int r = 0; r < 8; ++r) {
        int grow = mt * 16 + drow + r, gcol = nt * 16 + dcol;
        if (grow < 25) {
          float v = x[(size_t)n * 1600 + grow * 64 + gcol] + acc[r];
          xnew[(size_t)n * 1600 + grow * 64 + gcol] = v;
          sX[grow * 64 + gcol] = v;
        }
      }
    }
  __syncthreads();
  for (int c = lane; c < 64; c += 32) {
    float s = 0.f;
    for (int i = 0; i < 25; ++i) {
      float v = sX[i * 64 + c];
      s += v * v;
    }
    sInv[c] = rsqrtf(s / 25.f + 1e-6f) * gamma2[c];
  }
  __syncthreads();
  for (int q = lane; q < 200; q += 32) {  // 25*64/8 b128 chunks
    int base = q * 8;
    v8bf pk;
#pragma unroll
    for (int j = 0; j < 8; ++j) {
      int idx = base + j;
      pk[j] = (bf16)(sX[idx] * sInv[idx & 63]);
    }
    *(v8bf*)(h2b + (size_t)n * 1600 + base) = pk;
  }
  for (int o = lane; o < 128; o += 32) {
    float s = glb[o];
    for (int c = 0; c < 64; ++c) s += (sX[c] * sInv[c]) * glw[c * 128 + o];
    gaten[(size_t)n * 128 + o] = s;
  }
}

// ---------------- K7: node FFN with S2 activation (WMMA) ----------------
__global__ void __launch_bounds__(32)
k_node_ffn(const bf16* h2b, const float* gaten, const bf16* f1b, const bf16* f2b,
           const bf16* tgb, const bf16* fgb, const float* fb1, const float* fb2,
           const float* xnew, float* out) {
  const int n = blockIdx.x, lane = threadIdx.x;
  __shared__ __align__(16) bf16 sH[32 * 64];      // row-major A
  __shared__ __align__(16) bf16 sF1[32 * 128];    // swz, KB=2
  __shared__ __align__(16) bf16 sG[64 * 128];     // swz, KB=4
  __shared__ __align__(16) bf16 sF2in[32 * 128];  // row-major A
  for (int q = lane; q < 200; q += 32) {
    int base = q * 8;
    *(v8bf*)(sH + base) = *(const v8bf*)(h2b + (size_t)n * 1600 + base);
  }
  for (int q = lane; q < 7 * 8; q += 32)
    *(v8bf*)(sH + 25 * 64 + q * 8) = (v8bf)(bf16)0.f;
  for (int q = lane; q < 256; q += 32) {  // zero kh=3 blocks of sG
    int ct = q >> 5, w = q & 31;
    *(v8bf*)(sG + (ct * 4 + 3) * 256 + w * 8) = (v8bf)(bf16)0.f;
  }
  __syncthreads();
  const int drow = (lane & 16) ? 8 : 0, dcol = lane & 15;
  v8f zero = {};
  // f1 = h @ ffn_w1 (+b1 row0) -> sF1 (swz)
  for (int mt = 0; mt < 2; ++mt)
    for (int nt = 0; nt < 8; ++nt) {
      v8f acc = {};
      for (int kt = 0; kt < 2; ++kt) {
        v16bf A = load_a_rm(sH + mt * 16 * 64 + kt * 32, 64, lane);
        v16bf B = load_b_swz(f1b, 4, nt, kt, lane);
        acc = wmma_bf(A, B, acc);
      }
      v8f t;
#pragma unroll
      for (int r = 0; r < 8; ++r) {
        int grow = mt * 16 + drow + r, gcol = nt * 16 + dcol;
        float v = acc[r];
        if (grow == 0) v += fb1[gcol];
        if (grow >= 25) v = 0.f;
        t[r] = v;
      }
      store_d_swz(sF1, 2, nt, mt, lane, t);
    }
  __syncthreads();
  // grid = to_grid @ f1 ; silu -> sG (swz)
  for (int mt = 0; mt < 3; ++mt) {
    v16bf A = load_a_rm(tgb + mt * 16 * 32, 32, lane);
    for (int nt = 0; nt < 8; ++nt) {
      v16bf B = load_b_swz(sF1, 2, nt, 0, lane);
      v8f D = wmma_bf(A, B, zero);
      v8f t;
#pragma unroll
      for (int r = 0; r < 8; ++r) t[r] = siluf(D[r]);
      store_d_swz(sG, 4, nt, mt, lane, t);
    }
  }
  __syncthreads();
  // xg = from_grid @ grid ; row0 = silu(gate) -> sF2in (row-major A)
  for (int mt = 0; mt < 2; ++mt)
    for (int nt = 0; nt < 8; ++nt) {
      v8f acc = {};
      for (int kt = 0; kt < 2; ++kt) {
        v16bf A = load_a_rm(fgb + mt * 16 * 64 + kt * 32, 64, lane);
        v16bf B = load_b_swz(sG, 4, nt, kt, lane);
        acc = wmma_bf(A, B, acc);
      }
#pragma unroll
      for (int r = 0; r < 8; ++r) {
        int grow = mt * 16 + drow + r, gcol = nt * 16 + dcol;
        float v;
        if (grow == 0)      v = siluf(gaten[(size_t)n * 128 + gcol]);
        else if (grow < 25) v = acc[r];
        else                v = 0.f;
        sF2in[grow * 128 + gcol] = (bf16)v;
      }
    }
  __syncthreads();
  // f2 = f1' @ ffn_w2 (+b2 row0) ; out = xnew + f2
  for (int mt = 0; mt < 2; ++mt)
    for (int nt = 0; nt < 4; ++nt) {
      v8f acc = {};
      for (int kt = 0; kt < 4; ++kt) {
        v16bf A = load_a_rm(sF2in + mt * 16 * 128 + kt * 32, 128, lane);
        v16bf B = load_b_swz(f2b, 8, nt, kt, lane);
        acc = wmma_bf(A, B, acc);
      }
#pragma unroll
      for (int r = 0; r < 8; ++r) {
        int grow = mt * 16 + drow + r, gcol = nt * 16 + dcol;
        if (grow < 25) {
          float v = acc[r];
          if (grow == 0) v += fb2[gcol];
          out[(size_t)n * 1600 + grow * 64 + gcol] =
              xnew[(size_t)n * 1600 + grow * 64 + gcol] + v;
        }
      }
    }
}

extern "C" void kernel_launch(void* const* d_in, const int* in_sizes, int n_in,
                              void* d_out, int out_size, void* d_ws, size_t ws_size,
                              hipStream_t stream) {
  (void)in_sizes; (void)n_in; (void)out_size; (void)ws_size;
  const float* x      = (const float*)d_in[0];
  const int*   ei     = (const int*)d_in[1];
  const int*   anum   = (const int*)d_in[2];
  const float* edist  = (const float*)d_in[3];
  const float* wig    = (const float*)d_in[4];
  const float* stab   = (const float*)d_in[5];
  const float* ttab   = (const float*)d_in[6];
  const float* rw1    = (const float*)d_in[7];
  const float* rb1    = (const float*)d_in[8];
  const float* rw2    = (const float*)d_in[9];
  const float* rb2    = (const float*)d_in[10];
  const float* rw3    = (const float*)d_in[11];
  const float* rb3    = (const float*)d_in[12];
  const float* W1     = (const float*)d_in[13];
  const float* W0e    = (const float*)d_in[14];
  const float* adot   = (const float*)d_in[15];
  const float* tg     = (const float*)d_in[16];
  const float* fg     = (const float*)d_in[17];
  const float* proj   = (const float*)d_in[18];
  const float* gamma1 = (const float*)d_in[19];
  const float* gamma2 = (const float*)d_in[20];
  const float* glw    = (const float*)d_in[21];
  const float* glb    = (const float*)d_in[22];
  const float* fw1    = (const float*)d_in[23];
  const float* fb1    = (const float*)d_in[24];
  const float* fw2    = (const float*)d_in[25];
  const float* fb2    = (const float*)d_in[26];
  float* out = (float*)d_out;

  char* ws = (char*)d_ws;
  size_t off = 0;
  auto alloc = [&](size_t bytes) -> char* {
    char* p = ws + off;
    off = (off + bytes + 255) & ~(size_t)255;
    return p;
  };
  bf16* W1b   = (bf16*)alloc(128 * 64 * 2);
  bf16* tgb   = (bf16*)alloc(48 * 32 * 2);
  bf16* fgb   = (bf16*)alloc(32 * 64 * 2);
  bf16* projb = (bf16*)alloc(64 * 64 * 2);
  bf16* f1b   = (bf16*)alloc(64 * 128 * 2);
  bf16* f2b   = (bf16*)alloc(128 * 64 * 2);
  bf16* rw1b  = (bf16*)alloc(96 * 64 * 2);
  bf16* rw2b  = (bf16*)alloc(64 * 64 * 2);
  bf16* rw3b  = (bf16*)alloc(64 * 640 * 2);
  bf16* hbT   = (bf16*)alloc((size_t)NN * 2048 * 2);
  bf16* radb  = (bf16*)alloc((size_t)EE * 640 * 2);
  float* gate  = (float*)alloc((size_t)EE * 64 * 4);
  float* alpha = (float*)alloc((size_t)EE * 8 * 4);
  unsigned* amaxk = (unsigned*)alloc((size_t)NN * 8 * 4);
  float* asum  = (float*)alloc((size_t)NN * 8 * 4);
  float* agg   = (float*)alloc((size_t)NN * 1600 * 4);
  float* xnew  = (float*)alloc((size_t)NN * 1600 * 4);
  bf16* h2b    = (bf16*)alloc((size_t)NN * 1600 * 2);
  float* gaten = (float*)alloc((size_t)NN * 128 * 4);

  hipMemsetAsync(amaxk, 0, (size_t)NN * 8 * 4, stream);
  hipMemsetAsync(asum, 0, (size_t)NN * 8 * 4, stream);
  hipMemsetAsync(agg, 0, (size_t)NN * 1600 * 4, stream);

  k_prep<<<64, 256, 0, stream>>>(W1, tg, fg, proj, fw1, fw2, rw1, rw2, rw3, W1b,
                                 tgb, fgb, projb, f1b, f2b, rw1b, rw2b, rw3b);
  k_rms1<<<NN, 64, 0, stream>>>(x, gamma1, hbT);
  k_rad<<<EE / 16, 32, 0, stream>>>(ei, anum, edist, stab, ttab, rw1b, rb1, rw2b,
                                    rb2, rw3b, rb3, radb);
  k_alpha<<<EE, 32, 0, stream>>>(ei, wig, W0e, adot, hbT, radb, gate, alpha,
                                 amaxk);
  k_softsum<<<(EE * 8 + 255) / 256, 256, 0, stream>>>(ei, alpha, amaxk, asum);
  k_edge<<<EE, 32, 0, stream>>>(ei, wig, hbT, gate, alpha, amaxk, asum, radb, W1b,
                                tgb, fgb, agg);
  k_node_up<<<NN, 32, 0, stream>>>(x, agg, projb, gamma2, glw, glb, xnew, h2b,
                                   gaten);
  k_node_ffn<<<NN, 32, 0, stream>>>(h2b, gaten, f1b, f2b, tgb, fgb, fb1, fb2,
                                    xnew, out);
}
